// _BranchSIREN_25048249270690
// MI455X (gfx1250) — matrix operand order repaired and checked by reading, not compiled, not run on hardware
//
#include <hip/hip_runtime.h>

typedef __attribute__((ext_vector_type(16))) _Float16 v16h;
typedef __attribute__((ext_vector_type(8)))  _Float16 v8h;
typedef __attribute__((ext_vector_type(4)))  _Float16 v4h;
typedef __attribute__((ext_vector_type(8)))  float    v8f;
typedef int v4i __attribute__((vector_size(16)));   // matches builtin's parameter type

#define SW0    30.0f
#define WIDTH  512
#define NHID   6
#define DOUT   256
#define MROWS  128            // rows per workgroup
#define KC     32             // K per v_wmma_f32_16x16x32_f16
#define HS     520            // sH row stride in halfs (1040B -> bank stride 4, conflict-free)
#define WS     40             // panel row stride in halfs (80B -> bank stride 20, conflict-free)
#define PANELH (WIDTH * WS)   // 20480 halfs per panel buffer

#if defined(__gfx1250__) && __has_builtin(__builtin_amdgcn_global_load_async_to_lds_b128) && __has_builtin(__builtin_amdgcn_s_wait_asynccnt)
#define USE_ASYNC_LDS 1
#else
#define USE_ASYNC_LDS 0
#endif

extern __shared__ __align__(16) _Float16 smem[];

// stage rows x 32 k-columns of an f16 weight matrix into a fragment-ready LDS panel
__device__ __forceinline__ void stage_f16(_Float16* dst, const _Float16* src,
                                          int k0, int tid, int rows) {
    for (int v = tid; v < rows * 4; v += 256) {
        int row = v >> 2, q = v & 3;
        *(v8h*)(dst + row * WS + q * 8) =
            *(const v8h*)(src + (size_t)row * WIDTH + k0 + q * 8);
    }
}

// async variant: DMA straight into LDS (per-lane padded destinations), ASYNCcnt-tracked
__device__ __forceinline__ void stage_async(_Float16* dst, const _Float16* src,
                                            int k0, int tid, int rows) {
#if USE_ASYNC_LDS
    for (int v = tid; v < rows * 4; v += 256) {
        int row = v >> 2, q = v & 3;
        __builtin_amdgcn_global_load_async_to_lds_b128(
            (v4i*)(src + (size_t)row * WIDTH + k0 + q * 8),
            (v4i*)(dst + row * WS + q * 8),
            0, 0);
    }
#else
    stage_f16(dst, src, k0, tid, rows);
#endif
}

__device__ __forceinline__ void async_commit() {
#if USE_ASYNC_LDS
    __builtin_amdgcn_s_wait_asynccnt(0);
#endif
}

// fallback: stage while converting from f32 weights (no workspace available)
__device__ __forceinline__ void stage_f32(_Float16* dst, const float* src,
                                          int k0, int tid, int rows) {
    for (int v = tid; v < rows * 8; v += 256) {
        int row = v >> 3, kq = v & 7;
        float4 w4 = ((const float4*)(src + (size_t)row * WIDTH + k0))[kq];
        v4h h4 = { (_Float16)w4.x, (_Float16)w4.y, (_Float16)w4.z, (_Float16)w4.w };
        *(v4h*)(dst + row * WS + kq * 4) = h4;
    }
}

template <bool WF16>
__global__ __launch_bounds__(256)
void siren_wmma(const float* __restrict__ x,
                const float* __restrict__ W_in,  const float* __restrict__ b_in,
                const float* __restrict__ W_hid, const float* __restrict__ b_hid,
                const float* __restrict__ W_out, const float* __restrict__ b_out,
                const _Float16* __restrict__ wh16,   // f16 weights in workspace (or null)
                float* __restrict__ out)
{
    _Float16* sH  = smem;                       // MROWS x HS activations
    _Float16* sW0 = smem + MROWS * HS;          // panel buffer 0
    _Float16* sW1 = sW0 + PANELH;               // panel buffer 1

    const int tid  = threadIdx.x;
    const int lane = tid & 31;
    const int wave = tid >> 5;
    const int lmod = lane & 15;
    const int lhi  = lane >> 4;
    const int mg   = wave >> 2;                 // 0..1 : which 64-row half
    const int jg   = wave & 3;                  // 0..3 : which 128-col strip
    const long m0  = (long)blockIdx.x * MROWS;

    // kick off layer-1 weight panel DMA before doing any math
    if (WF16) stage_async(sW0, wh16, 0, tid, WIDTH);
    else      stage_f32(sW0, W_hid, 0, tid, WIDTH);

    // ---------------- layer 0: h = sin(W0*(x*W_in^T + b_in)) ----------------
    // (overlaps the panel DMA)
    for (int i = tid; i < MROWS * WIDTH; i += 256) {
        int m = i >> 9;
        int j = i & (WIDTH - 1);
        float z = SW0 * (x[m0 + m] * W_in[j] + b_in[j]);
        sH[m * HS + j] = (_Float16)__sinf(z);
    }
    async_commit();
    __syncthreads();

    // ---------------- hidden layers: h = sin(W0*(h@W^T + b)) ----------------
    for (int L = 0; L < NHID; ++L) {
        const _Float16* w16 = WF16 ? (wh16 + (size_t)L * WIDTH * WIDTH) : (const _Float16*)0;
        const float*    w32 = W_hid + (size_t)L * WIDTH * WIDTH;
        const float*    b   = b_hid + (size_t)L * WIDTH;

        v8f acc[4][8];
        #pragma unroll
        for (int mt = 0; mt < 4; ++mt)
            #pragma unroll
            for (int jt = 0; jt < 8; ++jt)
                acc[mt][jt] = (v8f){0.f,0.f,0.f,0.f,0.f,0.f,0.f,0.f};

        for (int kc = 0; kc < WIDTH / KC; ++kc) {
            _Float16* cur = (kc & 1) ? sW1 : sW0;
            _Float16* nxt = (kc & 1) ? sW0 : sW1;
            const int k0 = kc * KC;

            // kick off next panel: async DMA to LDS flies under the wmma chain
            if (kc < 15) {
                if (WF16) stage_async(nxt, w16, k0 + KC, tid, WIDTH);
                else      stage_f32(nxt, w32, k0 + KC, tid, WIDTH);
            }

            // A fragments: 4 M-tiles, each reused across 8 J-tiles
            v16h a[4];
            #pragma unroll
            for (int mt = 0; mt < 4; ++mt) {
                const _Float16* hr = sH + (mg * 64 + mt * 16 + lmod) * HS + k0;
                v8h alo = *(const v8h*)(hr + lhi * 16);
                v8h ahi = *(const v8h*)(hr + lhi * 16 + 8);
                #pragma unroll
                for (int t = 0; t < 8; ++t) { a[mt][t] = alo[t]; a[mt][t + 8] = ahi[t]; }
            }
            // B fragments: 8 J-tiles, each reused across 4 M-tiles
            v16h bf[8];
            #pragma unroll
            for (int jt = 0; jt < 8; ++jt)
                bf[jt] = *(const v16h*)(cur + ((jg * 8 + jt) * 16 + lmod) * WS + lhi * 16);

            #pragma unroll
            for (int jt = 0; jt < 8; ++jt)
                #pragma unroll
                for (int mt = 0; mt < 4; ++mt)
                    acc[mt][jt] = __builtin_amdgcn_wmma_f32_16x16x32_f16(
                        false, a[mt], false, bf[jt], (short)0, acc[mt][jt], false, false);

            if (WF16 && kc < 15) async_commit();
            __syncthreads();
        }

        // kick off the NEXT matrix's first panel (sW0 is free after the kc=15 barrier);
        // the DMA overlaps the sin epilogue below
        {
            const bool last = (L + 1 == NHID);
            if (WF16) {
                const _Float16* nw = wh16 + (size_t)(L + 1) * WIDTH * WIDTH; // W_out16 sits right after hidden
                stage_async(sW0, nw, 0, tid, last ? DOUT : WIDTH);
            } else {
                if (last) stage_f32(sW0, W_out, 0, tid, DOUT);
                else      stage_f32(sW0, W_hid + (size_t)(L + 1) * WIDTH * WIDTH, 0, tid, WIDTH);
            }
        }

        // epilogue: sin(W0*acc + W0*b) with the scale folded into one FMA
        #pragma unroll
        for (int jt = 0; jt < 8; ++jt) {
            int col = jg * 128 + jt * 16 + lmod;
            float w0b = SW0 * b[col];
            #pragma unroll
            for (int mt = 0; mt < 4; ++mt)
                #pragma unroll
                for (int r = 0; r < 8; ++r) {
                    int m = mg * 64 + mt * 16 + lhi * 8 + r;
                    sH[m * HS + col] =
                        (_Float16)__sinf(__builtin_fmaf(acc[mt][jt][r], SW0, w0b));
                }
        }

        async_commit();
        __syncthreads();
    }

    // ---------------- output layer: out = h@W_out^T + b_out (512 -> 256) ----
    // first panel already staged in sW0 under the last hidden epilogue
    {
        const _Float16* w16 = WF16 ? (wh16 + (size_t)NHID * WIDTH * WIDTH) : (const _Float16*)0;

        v8f acc[4][4];
        #pragma unroll
        for (int mt = 0; mt < 4; ++mt)
            #pragma unroll
            for (int jt = 0; jt < 4; ++jt)
                acc[mt][jt] = (v8f){0.f,0.f,0.f,0.f,0.f,0.f,0.f,0.f};

        for (int kc = 0; kc < WIDTH / KC; ++kc) {
            _Float16* cur = (kc & 1) ? sW1 : sW0;
            _Float16* nxt = (kc & 1) ? sW0 : sW1;
            const int k0 = kc * KC;

            if (kc < 15) {
                if (WF16) stage_async(nxt, w16, k0 + KC, tid, DOUT);
                else      stage_f32(nxt, W_out, k0 + KC, tid, DOUT);
            }

            v16h a[4];
            #pragma unroll
            for (int mt = 0; mt < 4; ++mt) {
                const _Float16* hr = sH + (mg * 64 + mt * 16 + lmod) * HS + k0;
                v8h alo = *(const v8h*)(hr + lhi * 16);
                v8h ahi = *(const v8h*)(hr + lhi * 16 + 8);
                #pragma unroll
                for (int t = 0; t < 8; ++t) { a[mt][t] = alo[t]; a[mt][t + 8] = ahi[t]; }
            }
            v16h bf[4];
            #pragma unroll
            for (int jt = 0; jt < 4; ++jt)
                bf[jt] = *(const v16h*)(cur + ((jg * 4 + jt) * 16 + lmod) * WS + lhi * 16);

            #pragma unroll
            for (int jt = 0; jt < 4; ++jt)
                #pragma unroll
                for (int mt = 0; mt < 4; ++mt)
                    acc[mt][jt] = __builtin_amdgcn_wmma_f32_16x16x32_f16(
                        false, a[mt], false, bf[jt], (short)0, acc[mt][jt], false, false);

            if (WF16 && kc < 15) async_commit();
            __syncthreads();
        }

        // stream f32 result to HBM (lanes 0..15 are 64B-contiguous per row)
        #pragma unroll
        for (int jt = 0; jt < 4; ++jt) {
            int col = jg * 64 + jt * 16 + lmod;
            float bj = b_out[col];
            #pragma unroll
            for (int mt = 0; mt < 4; ++mt)
                #pragma unroll
                for (int r = 0; r < 8; ++r) {
                    long m = m0 + mg * 64 + mt * 16 + lhi * 8 + r;
                    out[m * DOUT + col] = acc[mt][jt][r] + bj;
                }
        }
    }
}

// one-shot f32 -> f16 weight conversion into workspace ([W_hid | W_out])
__global__ __launch_bounds__(256)
void convert_w(const float* __restrict__ Wh, const float* __restrict__ Wo,
               _Float16* __restrict__ dst)
{
    const long NH4 = (long)NHID * WIDTH * WIDTH / 4;
    const long NT4 = NH4 + (long)DOUT * WIDTH / 4;
    long i = (long)blockIdx.x * 256 + threadIdx.x;
    if (i >= NT4) return;
    float4 w4 = (i < NH4) ? ((const float4*)Wh)[i] : ((const float4*)Wo)[i - NH4];
    v4h h4 = { (_Float16)w4.x, (_Float16)w4.y, (_Float16)w4.z, (_Float16)w4.w };
    ((v4h*)dst)[i] = h4;
}

extern "C" void kernel_launch(void* const* d_in, const int* in_sizes, int n_in,
                              void* d_out, int out_size, void* d_ws, size_t ws_size,
                              hipStream_t stream) {
    (void)n_in; (void)out_size;
    const float* x     = (const float*)d_in[0];
    const float* W_in  = (const float*)d_in[1];
    const float* b_in  = (const float*)d_in[2];
    const float* W_hid = (const float*)d_in[3];
    const float* b_hid = (const float*)d_in[4];
    const float* W_out = (const float*)d_in[5];
    const float* b_out = (const float*)d_in[6];
    float* out = (float*)d_out;

    const int N    = in_sizes[0];
    const int grid = N / MROWS;                                  // 2048 workgroups
    const size_t shmem = (size_t)(MROWS * HS + 2 * PANELH) * sizeof(_Float16); // ~210 KB

    const size_t wbytes = (size_t)(NHID * WIDTH * WIDTH + DOUT * WIDTH) * sizeof(_Float16);
    if (ws_size >= wbytes) {
        const long NT4 = ((long)NHID * WIDTH * WIDTH + (long)DOUT * WIDTH) / 4;
        convert_w<<<(int)((NT4 + 255) / 256), 256, 0, stream>>>(W_hid, W_out, (_Float16*)d_ws);
        siren_wmma<true><<<grid, 256, shmem, stream>>>(x, W_in, b_in, W_hid, b_hid,
                                                       W_out, b_out, (const _Float16*)d_ws, out);
    } else {
        siren_wmma<false><<<grid, 256, shmem, stream>>>(x, W_in, b_in, W_hid, b_hid,
                                                        W_out, b_out, (const _Float16*)0, out);
    }
}
